// ResidualLLRConnection_42709154792034
// MI455X (gfx1250) — matrix operand order repaired and checked by reading, not compiled, not run on hardware
//
#include <hip/hip_runtime.h>

typedef __attribute__((ext_vector_type(16))) _Float16 v16h;
typedef __attribute__((ext_vector_type(8)))  float    v8f;

#define Hdim        128
#define TILE_M      128      // rows per block
#define LDS_A_PITCH 132      // floats  (528B row pitch: 16B aligned, bank-shifted)
#define LDS_W_PITCH 136      // halves  (272B row pitch: 16B aligned, bank-shifted)

// Async global->LDS b128 (CDNA5, ASYNCcnt-tracked). lds_off = LDS byte address
// (low 32 bits of a generic shared pointer), gptr = global per-lane address.
__device__ __forceinline__ void async_ld_b128(unsigned lds_off, const void* gptr) {
  asm volatile("global_load_async_to_lds_b128 %0, %1, off"
               :: "v"(lds_off), "v"(gptr) : "memory");
}
__device__ __forceinline__ void wait_async0() {
  asm volatile("s_wait_asynccnt 0x0" ::: "memory");
}

// ---------------------------------------------------------------------------
// Prep: fold the rank-1 llr half of the GEMM into per-column u[], c[], and
// build an f16 transposed copy of W_gate[:128,:] (Wt[n][k]) in workspace.
// ws layout: float u[128] | float c[128] | _Float16 Wt[128*128]
// ---------------------------------------------------------------------------
__global__ void prep_kernel(const float* __restrict__ W_llr,
                            const float* __restrict__ b_llr,
                            const float* __restrict__ W_gate,
                            const float* __restrict__ b_gate,
                            float* __restrict__ ws) {
  const int j = threadIdx.x;          // 0..127 = output column
  float u = 0.0f, c = 0.0f;
  for (int k = 0; k < Hdim; ++k) {
    const float wg = W_gate[(size_t)(Hdim + k) * Hdim + j];
    u = fmaf(W_llr[k], wg, u);
    c = fmaf(b_llr[k], wg, c);
  }
  ws[j]        = u;
  ws[Hdim + j] = c + b_gate[j];

  _Float16* Wt = (_Float16*)(ws + 2 * Hdim);
  for (int k = 0; k < Hdim; ++k) {
    Wt[(size_t)j * Hdim + k] = (_Float16)W_gate[(size_t)k * Hdim + j];
  }
}

// ---------------------------------------------------------------------------
// Fused GEMM + sigmoid gate + blend for rows [0, V).
// One block = 128 rows, 8 waves; each wave owns a 16-row M tile and runs
// 8 N-tiles x 4 K-steps of v_wmma_f32_16x16x32_f16. Tiles staged with
// global_load_async_to_lds_b128.
// ---------------------------------------------------------------------------
__global__ __launch_bounds__(256) void fuse_kernel(
    const float* __restrict__ node_features,
    const float* __restrict__ llr,
    const float* __restrict__ W_llr,
    const float* __restrict__ b_llr,
    const float* __restrict__ ws,
    float* __restrict__ out,
    int V) {
  extern __shared__ char smem[];
  float*    sA  = (float*)smem;                                   // [128][132] f32
  _Float16* sW  = (_Float16*)(sA + TILE_M * LDS_A_PITCH);         // [128][136] f16
  float*    sU  = (float*)(sW + Hdim * LDS_W_PITCH);
  float*    sC  = sU + Hdim;
  float*    sWl = sC + Hdim;
  float*    sBl = sWl + Hdim;

  const int tid     = threadIdx.x;
  const int rowbase = blockIdx.x * TILE_M;

  // ---- stage W (f16, [n][k]) from workspace, async ------------------------
  const _Float16* Wt = (const _Float16*)(ws + 2 * Hdim);
  for (int f = tid; f < (Hdim * Hdim) / 8; f += 256) {  // 8 halves per b128
    const int n  = f >> 4;
    const int kq = (f & 15) << 3;
    async_ld_b128((unsigned)(uintptr_t)(sW + n * LDS_W_PITCH + kq),
                  Wt + (size_t)n * Hdim + kq);
  }
  if (tid < Hdim) {
    sU[tid]  = ws[tid];
    sC[tid]  = ws[Hdim + tid];
    sWl[tid] = W_llr[tid];
    sBl[tid] = b_llr[tid];
  }
  // ---- stage A tile (f32, async b128) + prefetch next tile ----------------
  for (int f = tid; f < TILE_M * (Hdim / 4); f += 256) {
    const int r   = f >> 5;
    const int cq  = (f & 31) << 2;
    const int row = rowbase + r;
    float* dstp = sA + r * LDS_A_PITCH + cq;
    if (row < V) {
      async_ld_b128((unsigned)(uintptr_t)dstp,
                    node_features + (size_t)row * Hdim + cq);
    } else {
      *(float4*)dstp = make_float4(0.f, 0.f, 0.f, 0.f);
    }
    const int prow = row + TILE_M;
    if (prow < V)
      __builtin_prefetch(node_features + (size_t)prow * Hdim + cq, 0, 1);
  }
  wait_async0();
  __syncthreads();

  const int wave  = tid >> 5;
  const int lane  = tid & 31;
  const int m0    = wave * 16;
  const int lhalf = lane >> 4;     // 0 | 1
  const int l15   = lane & 15;

  v8f acc[8] = {};

  const float*    pa  = sA + (m0 + l15) * LDS_A_PITCH;
  const _Float16* pb0 = sW + l15 * LDS_W_PITCH + lhalf * 8;

  #pragma unroll
  for (int kk = 0; kk < 4; ++kk) {
    const int kbase = kk * 32;
    // A fragment: lane<16 holds K {0..7,16..23}, lane>=16 holds {8..15,24..31}
    v16h a;
    {
      const int k0 = kbase + lhalf * 8;
      const float4 x0 = *(const float4*)(pa + k0);
      const float4 x1 = *(const float4*)(pa + k0 + 4);
      const float4 x2 = *(const float4*)(pa + k0 + 16);
      const float4 x3 = *(const float4*)(pa + k0 + 20);
      a[0]  = (_Float16)x0.x; a[1]  = (_Float16)x0.y;
      a[2]  = (_Float16)x0.z; a[3]  = (_Float16)x0.w;
      a[4]  = (_Float16)x1.x; a[5]  = (_Float16)x1.y;
      a[6]  = (_Float16)x1.z; a[7]  = (_Float16)x1.w;
      a[8]  = (_Float16)x2.x; a[9]  = (_Float16)x2.y;
      a[10] = (_Float16)x2.z; a[11] = (_Float16)x2.w;
      a[12] = (_Float16)x3.x; a[13] = (_Float16)x3.y;
      a[14] = (_Float16)x3.z; a[15] = (_Float16)x3.w;
    }
    // Preload all 8 B fragments of this K-step, then run the WMMA chain,
    // letting the scheduler hide ds latency under the matrix pipe.
    v16h bfrag[8];
    #pragma unroll
    for (int n = 0; n < 8; ++n) {
      const _Float16* pb = pb0 + n * 16 * LDS_W_PITCH + kbase;
      ((uint4*)&bfrag[n])[0] = *(const uint4*)(pb);
      ((uint4*)&bfrag[n])[1] = *(const uint4*)(pb + 16);
    }
    #pragma unroll
    for (int n = 0; n < 8; ++n) {
      acc[n] = __builtin_amdgcn_wmma_f32_16x16x32_f16(
          false, a, false, bfrag[n], (short)0, acc[n], false, false);
    }
  }

  // ---- epilogue: sigmoid gate + blend, x re-read from LDS -----------------
  const int mrow0 = rowbase + m0 + lhalf * 8;
  float llrv[8];
  #pragma unroll
  for (int r = 0; r < 8; ++r) {
    const int row = mrow0 + r;
    llrv[r] = (row < V) ? llr[row] : 0.0f;
  }
  #pragma unroll
  for (int n = 0; n < 8; ++n) {
    const int h   = n * 16 + l15;
    const float uh  = sU[h];
    const float ch  = sC[h];
    const float wlh = sWl[h];
    const float blh = sBl[h];
    #pragma unroll
    for (int r = 0; r < 8; ++r) {
      const int row = mrow0 + r;
      if (row >= V) continue;
      const float logit = acc[n][r] + llrv[r] * uh + ch;
      const float g     = 1.0f / (1.0f + __expf(-logit));
      const float lf    = llrv[r] * wlh + blh;
      const float vf    = sA[(m0 + lhalf * 8 + r) * LDS_A_PITCH + h];
      out[(size_t)row * Hdim + h] = g * lf + (1.0f - g) * vf;
    }
  }
}

// ---------------------------------------------------------------------------
// Passthrough copy for rows [V, N): pure streaming, float4.
// ---------------------------------------------------------------------------
__global__ void copy_kernel(const float4* __restrict__ src,
                            float4* __restrict__ dst, size_t n4) {
  size_t i       = (size_t)blockIdx.x * blockDim.x + threadIdx.x;
  const size_t s = (size_t)gridDim.x * blockDim.x;
  for (; i < n4; i += s) dst[i] = src[i];
}

extern "C" void kernel_launch(void* const* d_in, const int* in_sizes, int n_in,
                              void* d_out, int out_size, void* d_ws, size_t ws_size,
                              hipStream_t stream) {
  const float* node_features = (const float*)d_in[0];
  const float* llr           = (const float*)d_in[1];
  const float* W_llr         = (const float*)d_in[2];
  const float* b_llr         = (const float*)d_in[3];
  const float* W_gate        = (const float*)d_in[4];
  const float* b_gate        = (const float*)d_in[5];
  float* out = (float*)d_out;
  float* ws  = (float*)d_ws;

  const int       V  = in_sizes[1];
  const long long NH = in_sizes[0];          // N * H

  prep_kernel<<<1, Hdim, 0, stream>>>(W_llr, b_llr, W_gate, b_gate, ws);

  const int nblocks = (V + TILE_M - 1) / TILE_M;
  const size_t lds_bytes = (size_t)TILE_M * LDS_A_PITCH * sizeof(float)
                         + (size_t)Hdim * LDS_W_PITCH * sizeof(_Float16)
                         + (size_t)4 * Hdim * sizeof(float);
  fuse_kernel<<<nblocks, 256, lds_bytes, stream>>>(
      node_features, llr, W_llr, b_llr, ws, out, V);

  const size_t head  = (size_t)V * Hdim;
  const size_t tail4 = ((size_t)NH - head) / 4;
  if (tail4) {
    copy_kernel<<<2048, 256, 0, stream>>>(
        (const float4*)(node_features + head), (float4*)(out + head), tail4);
  }
}
